// MainModel_35837207117871
// MI455X (gfx1250) — compile-verified, matrix operand order checked by reading, not run on hardware
//
#include <hip/hip_runtime.h>

// ---------- types ----------
typedef __bf16 bf16_t;
typedef __attribute__((ext_vector_type(16))) __bf16 v16bf;
typedef __attribute__((ext_vector_type(8)))  float  v8f;

struct __attribute__((aligned(16))) U4 { unsigned int x, y, z, w; };

union FragBF {
    v16bf  v;
    U4     q[2];
    bf16_t e[16];
};

constexpr int CB = 8, CS = 1000, CD = 1024, CH = 16, CL = 4, CV = 32000;
constexpr int CM = CB * CS;            // 8000 token rows
constexpr float kInvSqrtDk = 0.125f;   // 1/sqrt(64)

static __device__ __forceinline__ v8f v8f_zero() {
    v8f z = {0.f, 0.f, 0.f, 0.f, 0.f, 0.f, 0.f, 0.f};
    return z;
}

static __device__ __forceinline__ v8f wmma_bf16(v16bf a, v16bf b, v8f c) {
    // D = A(16x32 bf16) x B(32x16 bf16) + C(16x16 f32)
    return __builtin_amdgcn_wmma_f32_16x16x32_bf16(false, a, false, b,
                                                   (short)0, c, false, false);
}

// CDNA5 async memory->LDS copy (ASYNCcnt tracked), ISA 15.18.3 op 98.
static __device__ __forceinline__ void async_copy_b128(bf16_t* lds_dst,
                                                       const bf16_t* gsrc) {
    const unsigned lds = (unsigned)(unsigned long long)(uintptr_t)lds_dst;
    asm volatile("global_load_async_to_lds_b128 %0, %1, off"
                 :: "v"(lds), "v"((unsigned long long)(uintptr_t)gsrc)
                 : "memory");
}
static __device__ __forceinline__ void wait_async0() {
    asm volatile("s_wait_asynccnt 0x0" ::: "memory");
}
static __device__ __forceinline__ void lds_fence() {
    asm volatile("s_wait_dscnt 0x0" ::: "memory");
}

// ---------- fp32 -> bf16 weight conversion ----------
__global__ void f2b_kernel(const float* __restrict__ src,
                           bf16_t* __restrict__ dst, long n) {
    long i = (long)blockIdx.x * blockDim.x + threadIdx.x;
    long stride = (long)gridDim.x * blockDim.x;
    for (; i < n; i += stride) dst[i] = (bf16_t)src[i];
}

// ---------- embedding + positional ----------
__global__ __launch_bounds__(256)
void embed_kernel(const int* __restrict__ x, const float* __restrict__ embed,
                  const float* __restrict__ pos,
                  float* __restrict__ hf, bf16_t* __restrict__ hb) {
    const long row = blockIdx.x;                 // b*CS + s
    const int  s   = (int)(row % CS);
    const long tok = x[row];
    for (int i = 0; i < 4; ++i) {
        const int c = threadIdx.x + i * 256;
        const float v = embed[tok * CD + c] + pos[(long)s * CD + c];
        hf[row * CD + c] = v;
        hb[row * CD + c] = (bf16_t)v;
    }
}

// ---------- bf16 GEMM: C[M,N] = A[M,K] * W[N,K]^T (+bias) ----------
// block tile 128x128, K-step 32, double-buffered LDS filled with
// global_load_async_to_lds_b128; 8 waves = 2(M) x 4(N); wave tile 64x32 =
// 4x2 v_wmma_f32_16x16x32_bf16.
// vtMode=1: bf16 output stored transposed per head as vT[B][H][64][S].
__global__ __launch_bounds__(256)
void gemm_bf16_kernel(const bf16_t* __restrict__ A, const bf16_t* __restrict__ W,
                      const float* __restrict__ bias,
                      float* __restrict__ Cf, bf16_t* __restrict__ Cb,
                      int M, int N, int K, int vtMode) {
    __shared__ bf16_t As[2][128][32];
    __shared__ bf16_t Bs[2][128][32];

    const int tid  = threadIdx.x;
    const int lane = tid & 31;
    const int wave = tid >> 5;
    const int wm   = wave >> 2;           // 0..1
    const int wn   = wave & 3;            // 0..3
    const int l16  = lane & 15;
    const int hl   = lane >> 4;
    const int kc   = hl * 8;              // per-lane K chunk base

    const long mBase = (long)blockIdx.y * 128;
    const long nBase = (long)blockIdx.x * 128;

    v8f acc[4][2];
    for (int t = 0; t < 4; ++t)
        for (int u = 0; u < 2; ++u) acc[t][u] = v8f_zero();

    // cooperative tile fill: 2 threads per row, 16 bf16 (2 x b128) each
    const int ldRow = tid >> 1;
    const int ldCol = (tid & 1) * 16;
    long ar = mBase + ldRow; if (ar >= M) ar = M - 1;   // clamped rows only
    long br = nBase + ldRow; if (br >= N) br = N - 1;   // feed unstored tiles
    const bf16_t* aPtr = A + ar * (long)K + ldCol;
    const bf16_t* wPtr = W + br * (long)K + ldCol;

    auto issueTile = [&](int buf, int k0) {
        async_copy_b128(&As[buf][ldRow][ldCol],     aPtr + k0);
        async_copy_b128(&As[buf][ldRow][ldCol + 8], aPtr + k0 + 8);
        async_copy_b128(&Bs[buf][ldRow][ldCol],     wPtr + k0);
        async_copy_b128(&Bs[buf][ldRow][ldCol + 8], wPtr + k0 + 8);
    };

    issueTile(0, 0);
    int buf = 0;
    for (int k0 = 0; k0 < K; k0 += 32) {
        wait_async0();                 // own 4 copies for this tile done
        __syncthreads();               // everyone's copies visible; prev reads done
        if (k0 + 32 < K) issueTile(buf ^ 1, k0 + 32);   // overlap with compute

        FragBF af[4], bw[2];
        #pragma unroll
        for (int t = 0; t < 4; ++t) {
            const int row = wm * 64 + t * 16 + l16;
            af[t].q[0] = *(const U4*)(&As[buf][row][kc]);
            af[t].q[1] = *(const U4*)(&As[buf][row][kc + 16]);
        }
        #pragma unroll
        for (int u = 0; u < 2; ++u) {
            const int row = wn * 32 + u * 16 + l16;
            bw[u].q[0] = *(const U4*)(&Bs[buf][row][kc]);
            bw[u].q[1] = *(const U4*)(&Bs[buf][row][kc + 16]);
        }
        #pragma unroll
        for (int t = 0; t < 4; ++t)
            #pragma unroll
            for (int u = 0; u < 2; ++u)
                acc[t][u] = wmma_bf16(af[t].v, bw[u].v, acc[t][u]);
        buf ^= 1;
    }

    // C layout: VGPR j -> row (j + 8*hl), col l16
    for (int u = 0; u < 2; ++u) {
        const long ncol = nBase + wn * 32 + u * 16 + l16;
        if (ncol >= N) continue;
        const float bval = bias ? bias[ncol] : 0.0f;
        for (int t = 0; t < 4; ++t) {
            const long mrow = mBase + wm * 64 + t * 16 + hl * 8;
            for (int j = 0; j < 8; ++j) {
                const long mr = mrow + j;
                if (mr >= M) break;
                const float v = acc[t][u][j] + bval;
                if (Cf) Cf[mr * (long)N + ncol] = v;
                if (Cb) {
                    if (vtMode) {   // vT[B][H][64][S], contraction dim contiguous
                        const long bb = mr / CS;
                        const int  ss = (int)(mr - bb * CS);
                        const long oidx =
                            ((bb * CH + (ncol >> 6)) * 64 + (ncol & 63)) * (long)CS + ss;
                        Cb[oidx] = (bf16_t)v;
                    } else {
                        Cb[mr * (long)N + ncol] = (bf16_t)v;
                    }
                }
            }
        }
    }
}

// ---------- flash attention: one wave per 16-row q-tile per (b,h) ----------
// Q,K in [B,S,D] bf16; V pre-transposed vT[B][H][64][S] so P*V B-fragments
// are two aligned b128 global loads per dk tile. Waves are fully independent
// (per-wave LDS slice for the P relayout; DS ops are in-order per wave).
__global__ __launch_bounds__(256)
void attn_kernel(const bf16_t* __restrict__ Q, const bf16_t* __restrict__ Kp,
                 const bf16_t* __restrict__ Vt, bf16_t* __restrict__ O) {
    __shared__ bf16_t Pl[8][16][32];   // per-wave P tile (A-layout staging)

    const int lane = threadIdx.x & 31;
    const int wave = threadIdx.x >> 5;
    const int l16  = lane & 15;
    const int hl   = lane >> 4;
    const int kc   = hl * 8;

    const int bid   = blockIdx.x;      // = b*128 + h*8 + chunk
    const int chunk = bid & 7;
    const int h     = (bid >> 3) & (CH - 1);
    const int b     = bid >> 7;
    int qt = chunk * 8 + wave;         // 63 q-tiles of 16 rows (0..62)
    if (qt > 62) qt = 62;              // duplicate last tile (benign rewrite)

    int qrow = qt * 16 + l16;
    if (qrow >= CS) qrow = CS - 1;     // clamped rows are never stored
    const long qoff = ((long)b * CS + qrow) * CD + h * 64;

    FragBF qf[2];                      // Q 16x64 as two 16x32 A-frags
    #pragma unroll
    for (int t = 0; t < 2; ++t) {
        qf[t].q[0] = *(const U4*)(Q + qoff + t * 32 + kc);
        qf[t].q[1] = *(const U4*)(Q + qoff + t * 32 + kc + 16);
    }

    // per-lane V row base: lane owns dk column (t*16 + l16) of each tile
    const bf16_t* vrow[4];
    #pragma unroll
    for (int t = 0; t < 4; ++t)
        vrow[t] = Vt + (((long)b * CH + h) * 64 + (t * 16 + l16)) * (long)CS;

    v8f oacc[4];
    for (int t = 0; t < 4; ++t) oacc[t] = v8f_zero();
    float mstat[8], lsum[8];
    for (int j = 0; j < 8; ++j) { mstat[j] = -3.0e38f; lsum[j] = 0.f; }

    for (int kb = 0; kb < 32; ++kb) {          // 32 keys/iter, 1024 >= S
        // two 16-key score tiles: S = Q K^T / sqrt(dk)
        float sv[2][8];
        float rowmax[8];
        for (int j = 0; j < 8; ++j) rowmax[j] = -3.0e38f;
        for (int sub = 0; sub < 2; ++sub) {
            const int kcol = kb * 32 + sub * 16 + l16;
            const int kr   = kcol < CS ? kcol : CS - 1;
            const long koff = ((long)b * CS + kr) * CD + h * 64;
            FragBF kf0, kf1;
            kf0.q[0] = *(const U4*)(Kp + koff + kc);
            kf0.q[1] = *(const U4*)(Kp + koff + kc + 16);
            kf1.q[0] = *(const U4*)(Kp + koff + 32 + kc);
            kf1.q[1] = *(const U4*)(Kp + koff + 32 + kc + 16);
            v8f sc = v8f_zero();
            sc = wmma_bf16(qf[0].v, kf0.v, sc);
            sc = wmma_bf16(qf[1].v, kf1.v, sc);
            const bool valid = (kcol < CS);
            for (int j = 0; j < 8; ++j) {
                const float s = valid ? sc[j] * kInvSqrtDk : -1.0e30f;
                sv[sub][j] = s;
                rowmax[j]  = fmaxf(rowmax[j], s);
            }
        }
        // row max across the 16 lanes of each half-wave
        for (int j = 0; j < 8; ++j) {
            float tv = rowmax[j];
            for (int i = 1; i < 16; i <<= 1) tv = fmaxf(tv, __shfl_xor(tv, i, 16));
            rowmax[j] = tv;
        }
        // online softmax update
        float p[2][8];
        for (int j = 0; j < 8; ++j) {
            const float mnew = fmaxf(mstat[j], rowmax[j]);
            const float corr = __expf(mstat[j] - mnew);
            mstat[j] = mnew;
            const float p0 = __expf(sv[0][j] - mnew);
            const float p1 = __expf(sv[1][j] - mnew);
            p[0][j] = p0; p[1][j] = p1;
            float rs = p0 + p1;
            for (int i = 1; i < 16; i <<= 1) rs += __shfl_xor(rs, i, 16);
            lsum[j] = lsum[j] * corr + rs;
            for (int t = 0; t < 4; ++t) oacc[t][j] *= corr;
        }
        // P (C layout) -> per-wave LDS 16x32 row-major for A-frag re-layout
        for (int j = 0; j < 8; ++j) {
            Pl[wave][hl * 8 + j][l16]      = (bf16_t)p[0][j];
            Pl[wave][hl * 8 + j][16 + l16] = (bf16_t)p[1][j];
        }
        lds_fence();                   // DS in-order per wave; wait stores done

        FragBF pf;
        pf.q[0] = *(const U4*)(&Pl[wave][l16][kc]);
        pf.q[1] = *(const U4*)(&Pl[wave][l16][kc + 16]);
        #pragma unroll
        for (int t = 0; t < 4; ++t) {  // O += P * V, dk in 4 N-tiles
            FragBF vf;
            vf.q[0] = *(const U4*)(vrow[t] + kb * 32 + kc);
            vf.q[1] = *(const U4*)(vrow[t] + kb * 32 + kc + 16);
            oacc[t] = wmma_bf16(pf.v, vf.v, oacc[t]);
        }
    }

    // normalize and store concat-head layout [B,S,D]
    for (int j = 0; j < 8; ++j) {
        const int row = qt * 16 + hl * 8 + j;
        if (row >= CS) continue;
        const float inv = 1.0f / lsum[j];
        const long ooff = ((long)b * CS + row) * CD + h * 64;
        for (int t = 0; t < 4; ++t)
            O[ooff + t * 16 + l16] = (bf16_t)(oacc[t][j] * inv);
    }
}

// ---------- residual add + LayerNorm ----------
__global__ __launch_bounds__(256)
void add_ln_kernel(const float* __restrict__ a, const float* __restrict__ res,
                   const float* __restrict__ gamma, const float* __restrict__ beta,
                   float* __restrict__ hf, bf16_t* __restrict__ hb) {
    __shared__ float red[256];
    const int  tid  = threadIdx.x;
    const long base = (long)blockIdx.x * CD;

    float v[4];
    float s = 0.f;
    for (int i = 0; i < 4; ++i) {
        const int c = tid + i * 256;
        v[i] = a[base + c] + res[base + c];
        s += v[i];
    }
    red[tid] = s; __syncthreads();
    for (int off = 128; off > 0; off >>= 1) {
        if (tid < off) red[tid] += red[tid + off];
        __syncthreads();
    }
    const float mean = red[0] * (1.0f / CD);
    __syncthreads();

    float s2 = 0.f;
    for (int i = 0; i < 4; ++i) { const float d = v[i] - mean; s2 += d * d; }
    red[tid] = s2; __syncthreads();
    for (int off = 128; off > 0; off >>= 1) {
        if (tid < off) red[tid] += red[tid + off];
        __syncthreads();
    }
    const float rstd = rsqrtf(red[0] * (1.0f / CD) + 1e-5f);

    for (int i = 0; i < 4; ++i) {
        const int c = tid + i * 256;
        const float o = (v[i] - mean) * rstd * gamma[c] + beta[c];
        hf[base + c] = o;
        hb[base + c] = (bf16_t)o;
    }
}

// ---------- launch ----------
extern "C" void kernel_launch(void* const* d_in, const int* in_sizes, int n_in,
                              void* d_out, int out_size, void* d_ws, size_t ws_size,
                              hipStream_t stream) {
    (void)in_sizes; (void)n_in; (void)out_size; (void)ws_size;

    const int*   x     = (const int*)d_in[0];
    const float* embed = (const float*)d_in[1];
    const float* pos   = (const float*)d_in[2];
    const float* Wq    = (const float*)d_in[3];
    const float* bq    = (const float*)d_in[4];
    const float* Wk    = (const float*)d_in[5];
    const float* bk    = (const float*)d_in[6];
    const float* Wv    = (const float*)d_in[7];
    const float* bv    = (const float*)d_in[8];
    const float* Wo    = (const float*)d_in[9];
    const float* bo    = (const float*)d_in[10];
    const float* gamma = (const float*)d_in[11];
    const float* beta  = (const float*)d_in[12];
    const float* Wf    = (const float*)d_in[13];
    const float* bfv   = (const float*)d_in[14];
    float* out = (float*)d_out;

    char* w = (char*)d_ws;
    auto take = [&](size_t bytes) -> char* {
        char* p = w; w += (bytes + 255) & ~(size_t)255; return p;
    };
    const size_t szW   = (size_t)CL * CD * CD;  // per-proj weight elems (all layers)
    const size_t szWf  = (size_t)CV * CD;
    const size_t szAct = (size_t)CB * CS * CD;

    bf16_t* WqB = (bf16_t*)take(szW * 2);
    bf16_t* WkB = (bf16_t*)take(szW * 2);
    bf16_t* WvB = (bf16_t*)take(szW * 2);
    bf16_t* WoB = (bf16_t*)take(szW * 2);
    bf16_t* WfB = (bf16_t*)take(szWf * 2);
    float*  hF  = (float*) take(szAct * 4);
    bf16_t* hB  = (bf16_t*)take(szAct * 2);
    bf16_t* qB  = (bf16_t*)take(szAct * 2);
    bf16_t* kB  = (bf16_t*)take(szAct * 2);
    bf16_t* vT  = (bf16_t*)take(szAct * 2 + 256);   // +pad for tail overread
    bf16_t* aB  = (bf16_t*)take(szAct * 2);
    float*  aoF = (float*) take(szAct * 4);

    f2b_kernel<<<2048, 256, 0, stream>>>(Wq, WqB, (long)szW);
    f2b_kernel<<<2048, 256, 0, stream>>>(Wk, WkB, (long)szW);
    f2b_kernel<<<2048, 256, 0, stream>>>(Wv, WvB, (long)szW);
    f2b_kernel<<<2048, 256, 0, stream>>>(Wo, WoB, (long)szW);
    f2b_kernel<<<4096, 256, 0, stream>>>(Wf, WfB, (long)szWf);
    embed_kernel<<<CB * CS, 256, 0, stream>>>(x, embed, pos, hF, hB);

    const dim3 gProj((CD + 127) / 128, (CM + 127) / 128);   // (8, 63)
    for (int l = 0; l < CL; ++l) {
        const size_t wo = (size_t)l * CD * CD;
        gemm_bf16_kernel<<<gProj, 256, 0, stream>>>(hB, WqB + wo, bq + l * CD,
                                                    nullptr, qB, CM, CD, CD, 0);
        gemm_bf16_kernel<<<gProj, 256, 0, stream>>>(hB, WkB + wo, bk + l * CD,
                                                    nullptr, kB, CM, CD, CD, 0);
        gemm_bf16_kernel<<<gProj, 256, 0, stream>>>(hB, WvB + wo, bv + l * CD,
                                                    nullptr, vT, CM, CD, CD, 1);
        attn_kernel<<<CB * CH * 8, 256, 0, stream>>>(qB, kB, vT, aB);
        gemm_bf16_kernel<<<gProj, 256, 0, stream>>>(aB, WoB + wo, bo + l * CD,
                                                    aoF, nullptr, CM, CD, CD, 0);
        add_ln_kernel<<<CB * CS, 256, 0, stream>>>(aoF, hF, gamma, beta, hF, hB);
    }

    const dim3 gOut((CV + 127) / 128, (CM + 127) / 128);    // (250, 63)
    gemm_bf16_kernel<<<gOut, 256, 0, stream>>>(hB, WfB, bfv, out, nullptr,
                                               CM, CV, CD, 0);
}
